// HoppingBlock_71055938945439
// MI455X (gfx1250) — matrix operand order repaired and checked by reading, not compiled, not run on hardware
//
#include <hip/hip_runtime.h>
#include <hip/hip_bf16.h>

typedef __attribute__((ext_vector_type(16))) _Float16 v16h;
typedef __attribute__((ext_vector_type(8)))  float    v8f;

// ---------------------------------------------------------------------------
// WMMA GEMM: C[M,N] = act(A[M,K](f32) @ B[K,N] + bias), B given as f16
// transposed Bt[N,K].  One wave computes a 16x32 tile of C (two 16x16 WMMA
// n-tiles sharing one A fragment).  K,N compile-time (N multiple of 32,
// K multiple of 32); M runtime, tail-guarded.  ACT: 0=none,1=relu,2=lrelu.01
// ---------------------------------------------------------------------------
template <int K, int N, int ACT>
__global__ __launch_bounds__(128) void k_gemm(
    const float* __restrict__ A, const _Float16* __restrict__ Bt,
    const float* __restrict__ bias, float* __restrict__ C, int M)
{
  const int lane = threadIdx.x & 31;
  const int wave = threadIdx.x >> 5;
  const int mbase = (blockIdx.y * 4 + wave) * 16;
  if (mbase >= M) return;                 // wave-uniform exit: EXEC all-1 below
  const int nbase = blockIdx.x * 32;
  const int hl  = lane >> 4;              // half-wave select
  const int l15 = lane & 15;

  int arow = mbase + l15; if (arow >= M) arow = M - 1;   // clamp, guard on store
  const float*    ap0  = A  + (size_t)arow * K + (hl ? 16 : 0);
  const _Float16* bp0a = Bt + (size_t)(nbase + l15) * K + (hl ? 16 : 0);
  const _Float16* bp0b = bp0a + (size_t)16 * K;

  v8f acc0 = {}, acc1 = {};
#pragma unroll
  for (int k0 = 0; k0 < K; k0 += 32) {
    const float* ap = ap0 + k0;
    __builtin_prefetch(ap + 32, 0, 1);    // gfx1250 global_prefetch_b8
    // A 16x32 f16 fragment: lanes 0-15 hold K 0..15, lanes 16-31 hold K 16..31
    v16h a;
#pragma unroll
    for (int t = 0; t < 8; ++t) a[t]     = (_Float16)ap[t];
#pragma unroll
    for (int t = 0; t < 8; ++t) a[8 + t] = (_Float16)ap[8 + t];
    // B 32x16 f16 fragments: lanes 0-15 hold K 0..15, lanes 16-31 hold K 16..31
    const _Float16* bpa = bp0a + k0;
    const _Float16* bpb = bp0b + k0;
    v16h b0, b1;
#pragma unroll
    for (int t = 0; t < 16; ++t) b0[t] = bpa[t];
#pragma unroll
    for (int t = 0; t < 16; ++t) b1[t] = bpb[t];
    acc0 = __builtin_amdgcn_wmma_f32_16x16x32_f16(false, a, false, b0,
                                                  (short)0, acc0, false, false);
    acc1 = __builtin_amdgcn_wmma_f32_16x16x32_f16(false, a, false, b1,
                                                  (short)0, acc1, false, false);
  }
  // C layout: lane<16 -> col base+lane rows mbase+0..7 ; lane>=16 -> rows +8..15
  const int col0  = nbase + l15;
  const int col1  = col0 + 16;
  const int rbase = mbase + (hl ? 8 : 0);
  const float bv0 = bias ? bias[col0] : 0.0f;
  const float bv1 = bias ? bias[col1] : 0.0f;
  float* __restrict__ c0 = C + (size_t)rbase * N + col0;
  float* __restrict__ c1 = C + (size_t)rbase * N + col1;
  if (mbase + 16 <= M) {
    // full tile: unguarded stores
#pragma unroll
    for (int r = 0; r < 8; ++r) {
      float v0 = acc0[r] + bv0, v1 = acc1[r] + bv1;
      if (ACT == 1) { v0 = fmaxf(v0, 0.0f); v1 = fmaxf(v1, 0.0f); }
      if (ACT == 2) { v0 = (v0 >= 0.0f) ? v0 : 0.01f * v0;
                      v1 = (v1 >= 0.0f) ? v1 : 0.01f * v1; }
      c0[(size_t)r * N] = v0;
      c1[(size_t)r * N] = v1;
    }
  } else {
#pragma unroll
    for (int r = 0; r < 8; ++r) {
      if (rbase + r < M) {
        float v0 = acc0[r] + bv0, v1 = acc1[r] + bv1;
        if (ACT == 1) { v0 = fmaxf(v0, 0.0f); v1 = fmaxf(v1, 0.0f); }
        if (ACT == 2) { v0 = (v0 >= 0.0f) ? v0 : 0.01f * v0;
                        v1 = (v1 >= 0.0f) ? v1 : 0.01f * v1; }
        c0[(size_t)r * N] = v0;
        c1[(size_t)r * N] = v1;
      }
    }
  }
}

// ---------------------------------------------------------------------------
// Small helper kernels
// ---------------------------------------------------------------------------
__global__ void k_fill(float* __restrict__ p, float v, int n) {
  int i = blockIdx.x * blockDim.x + threadIdx.x;
  if (i < n) p[i] = v;
}

// f16 transposed weight: Bt[n*K+k] = W[k*N+n]
__global__ void k_convw(const float* __restrict__ W, _Float16* __restrict__ Bt,
                        int K, int N) {
  int i = blockIdx.x * blockDim.x + threadIdx.x;
  if (i >= K * N) return;
  int n = i / K, k = i - n * K;
  Bt[i] = (_Float16)W[(size_t)k * N + n];
}

// fold g1_We[64,256] with g1_att_e[4,64] -> Wp[64,4]
__global__ void k_fold_atte(const float* __restrict__ We,
                            const float* __restrict__ att,
                            float* __restrict__ Wp) {
  int i = blockIdx.x * blockDim.x + threadIdx.x;
  if (i >= 64 * 4) return;
  int k = i >> 2, hh = i & 3;
  float s = 0.0f;
  for (int c = 0; c < 64; ++c) s += We[k * 256 + hh * 64 + c] * att[hh * 64 + c];
  Wp[i] = s;
}

// Y[M,N] = X[M,K] @ W[K,N] (tiny N, scalar)
__global__ void k_gemv(const float* __restrict__ X, const float* __restrict__ W,
                       float* __restrict__ Y, int M, int K, int N) {
  int i = blockIdx.x * blockDim.x + threadIdx.x;
  if (i >= M * N) return;
  int m = i / N, n = i - m * N;
  float s = 0.0f;
  for (int k = 0; k < K; ++k) s += X[(size_t)m * K + k] * W[k * N + n];
  Y[i] = s;
}

// asrc/adst per edge+head from xl [E,256] and att vectors [4,64]
__global__ void k_attfold(const float* __restrict__ xl,
                          const float* __restrict__ att_s,
                          const float* __restrict__ att_d,
                          float* __restrict__ asrc, float* __restrict__ adst, int E) {
  int i = blockIdx.x * blockDim.x + threadIdx.x;
  if (i >= E * 4) return;
  int e = i >> 2, hh = i & 3;
  const float* xr = xl + (size_t)e * 256 + hh * 64;
  float s1 = 0.0f, s2 = 0.0f;
  for (int c = 0; c < 64; ++c) {
    float v = xr[c];
    s1 += v * att_s[hh * 64 + c];
    s2 += v * att_d[hh * 64 + c];
  }
  asrc[i] = s1; adst[i] = s2;
}

// in-degree + self-loop attr (head projection) accumulation
__global__ void k_deg_sl1(const int* __restrict__ ls, const int* __restrict__ ld,
                          const int* __restrict__ edst, const float* __restrict__ xWe1,
                          float* __restrict__ deg, float* __restrict__ sl1, int LE) {
  int l = blockIdx.x * blockDim.x + threadIdx.x;
  if (l >= LE) return;
  int e = ld[l], j = edst[ls[l]];
  atomicAdd(&deg[e], 1.0f);
  for (int hh = 0; hh < 4; ++hh) atomicAdd(&sl1[e * 4 + hh], xWe1[j * 4 + hh]);
}

__global__ void k_sl2(const int* __restrict__ ls, const int* __restrict__ ld,
                      const int* __restrict__ edst, const float* __restrict__ xWe2,
                      float* __restrict__ sl2, int LE) {
  int i = blockIdx.x * blockDim.x + threadIdx.x;
  if (i >= LE * 64) return;
  int l = i >> 6, c = i & 63;
  int e = ld[l], j = edst[ls[l]];
  atomicAdd(&sl2[(size_t)e * 64 + c], xWe2[(size_t)j * 64 + c]);
}

__global__ void k_dinv(const float* __restrict__ deg, float* __restrict__ dinv, int n) {
  int e = blockIdx.x * blockDim.x + threadIdx.x;
  if (e >= n) return;
  float d = deg[e];
  dinv[e] = d > 0.0f ? rsqrtf(fmaxf(d, 1.0f)) : 0.0f;
}

// ---- GAT layer 1 (4 heads) ----
__global__ void k_gat1_self(const float* __restrict__ asrc, const float* __restrict__ adst,
                            const float* __restrict__ sl1, const float* __restrict__ deg,
                            float* __restrict__ aself, float* __restrict__ segsum, int E) {
  int i = blockIdx.x * blockDim.x + threadIdx.x;
  if (i >= E * 4) return;
  int e = i >> 2;
  float inv = 1.0f / fmaxf(deg[e], 1.0f);
  float a = asrc[i] + adst[i] + sl1[i] * inv;
  a = (a >= 0.0f) ? a : 0.2f * a;
  float ex = __expf(a);
  aself[i] = ex;
  segsum[i] = ex;               // init denominator with self-loop term
}

__global__ void k_gat1_line(const int* __restrict__ ls, const int* __restrict__ ld,
                            const int* __restrict__ edst,
                            const float* __restrict__ asrc, const float* __restrict__ adst,
                            const float* __restrict__ xWe1,
                            float* __restrict__ alphaL, float* __restrict__ segsum, int LE) {
  int i = blockIdx.x * blockDim.x + threadIdx.x;
  if (i >= LE * 4) return;
  int l = i >> 2, hh = i & 3;
  int s = ls[l], d = ld[l], j = edst[s];
  float a = asrc[s * 4 + hh] + adst[d * 4 + hh] + xWe1[j * 4 + hh];
  a = (a >= 0.0f) ? a : 0.2f * a;
  float ex = __expf(a);
  alphaL[i] = ex;
  atomicAdd(&segsum[d * 4 + hh], ex);
}

__global__ void k_gat1_init(const float* __restrict__ aself, const float* __restrict__ segsum,
                            const float* __restrict__ xl, const float* __restrict__ b,
                            float* __restrict__ h, int E) {
  int i = blockIdx.x * blockDim.x + threadIdx.x;
  if (i >= E * 64) return;
  int e = i >> 6, c = i & 63;
  float acc = 0.0f;
  for (int hh = 0; hh < 4; ++hh)
    acc += (aself[e * 4 + hh] / (segsum[e * 4 + hh] + 1e-16f)) *
           xl[(size_t)e * 256 + hh * 64 + c];
  h[i] = acc * 0.25f + b[c];
}

__global__ void k_gat1_agg(const int* __restrict__ ls, const int* __restrict__ ld,
                           const float* __restrict__ alphaL, const float* __restrict__ segsum,
                           const float* __restrict__ xl, float* __restrict__ h, int LE) {
  int i = blockIdx.x * blockDim.x + threadIdx.x;
  if (i >= LE * 64) return;
  int l = i >> 6, c = i & 63;
  int s = ls[l], d = ld[l];
  float acc = 0.0f;
  for (int hh = 0; hh < 4; ++hh)
    acc += (alphaL[l * 4 + hh] / (segsum[d * 4 + hh] + 1e-16f)) *
           xl[(size_t)s * 256 + hh * 64 + c];
  atomicAdd(&h[(size_t)d * 64 + c], acc * 0.25f);
}

// ---- GATv2 layer ----
__global__ void k_gat2_self(const float* __restrict__ xlv, const float* __restrict__ xrv,
                            const float* __restrict__ sl2, const float* __restrict__ deg,
                            const float* __restrict__ att,
                            float* __restrict__ aself, float* __restrict__ segsum, int E) {
  int e = blockIdx.x * blockDim.x + threadIdx.x;
  if (e >= E) return;
  float inv = 1.0f / fmaxf(deg[e], 1.0f);
  float t = 0.0f;
  for (int c = 0; c < 64; ++c) {
    float v = xlv[(size_t)e * 64 + c] + xrv[(size_t)e * 64 + c] + sl2[(size_t)e * 64 + c] * inv;
    v = (v >= 0.0f) ? v : 0.2f * v;
    t += v * att[c];
  }
  float ex = __expf(t);
  aself[e] = ex; segsum[e] = ex;
}

__global__ void k_gat2_line(const int* __restrict__ ls, const int* __restrict__ ld,
                            const int* __restrict__ edst,
                            const float* __restrict__ xlv, const float* __restrict__ xrv,
                            const float* __restrict__ xWe2, const float* __restrict__ att,
                            float* __restrict__ alphaL, float* __restrict__ segsum, int LE) {
  int l = blockIdx.x * blockDim.x + threadIdx.x;
  if (l >= LE) return;
  int s = ls[l], d = ld[l], j = edst[s];
  float t = 0.0f;
  for (int c = 0; c < 64; ++c) {
    float v = xlv[(size_t)s * 64 + c] + xrv[(size_t)d * 64 + c] + xWe2[(size_t)j * 64 + c];
    v = (v >= 0.0f) ? v : 0.2f * v;
    t += v * att[c];
  }
  float ex = __expf(t);
  alphaL[l] = ex;
  atomicAdd(&segsum[d], ex);
}

__global__ void k_gat2_init(const float* __restrict__ aself, const float* __restrict__ segsum,
                            const float* __restrict__ xlv, const float* __restrict__ b,
                            float* __restrict__ h2, int E) {
  int i = blockIdx.x * blockDim.x + threadIdx.x;
  if (i >= E * 64) return;
  int e = i >> 6, c = i & 63;
  h2[i] = (aself[e] / (segsum[e] + 1e-16f)) * xlv[i] + b[c];
}

__global__ void k_gat2_agg(const int* __restrict__ ls, const int* __restrict__ ld,
                           const float* __restrict__ alphaL, const float* __restrict__ segsum,
                           const float* __restrict__ xlv, float* __restrict__ h2, int LE) {
  int i = blockIdx.x * blockDim.x + threadIdx.x;
  if (i >= LE * 64) return;
  int l = i >> 6, c = i & 63;
  int s = ls[l], d = ld[l];
  atomicAdd(&h2[(size_t)d * 64 + c],
            (alphaL[l] / (segsum[d] + 1e-16f)) * xlv[(size_t)s * 64 + c]);
}

// ---- ARMA aggregation ----
__global__ void k_arma_agg(const int* __restrict__ ls, const int* __restrict__ ld,
                           const float* __restrict__ dinv, const float* __restrict__ hWi,
                           float* __restrict__ out, int LE) {
  int i = blockIdx.x * blockDim.x + threadIdx.x;
  if (i >= LE * 64) return;
  int l = i >> 6, c = i & 63;
  int s = ls[l], d = ld[l];
  atomicAdd(&out[(size_t)d * 64 + c], dinv[s] * dinv[d] * hWi[(size_t)s * 64 + c]);
}

__global__ void k_relu(float* __restrict__ p, int n) {
  int i = blockIdx.x * blockDim.x + threadIdx.x;
  if (i < n) p[i] = fmaxf(p[i], 0.0f);
}

// ---- GINE ----
__global__ void k_zinit(const float* __restrict__ h, const float* __restrict__ epsp,
                        float* __restrict__ z, int n) {
  int i = blockIdx.x * blockDim.x + threadIdx.x;
  if (i < n) z[i] = (1.0f + epsp[0]) * h[i];
}

__global__ void k_gine_agg(const int* __restrict__ ls, const int* __restrict__ ld,
                           const int* __restrict__ edst,
                           const float* __restrict__ h, const float* __restrict__ xWeg,
                           const float* __restrict__ be, float* __restrict__ z, int LE) {
  int i = blockIdx.x * blockDim.x + threadIdx.x;
  if (i >= LE * 64) return;
  int l = i >> 6, c = i & 63;
  int s = ls[l], d = ld[l], j = edst[s];
  float m = h[(size_t)s * 64 + c] + xWeg[(size_t)j * 64 + c] + be[c];
  atomicAdd(&z[(size_t)d * 64 + c], fmaxf(m, 0.0f));
}

// ---------------------------------------------------------------------------
extern "C" void kernel_launch(void* const* d_in, const int* in_sizes, int n_in,
                              void* d_out, int out_size, void* d_ws, size_t ws_size,
                              hipStream_t stream) {
  const float* x        = (const float*)d_in[0];
  const int*   ei       = (const int*)  d_in[1];
  const float* eattr    = (const float*)d_in[2];
  const int*   ls       = (const int*)  d_in[3];
  const int*   ld       = (const int*)  d_in[4];
  const float* g1_W     = (const float*)d_in[5];
  const float* g1_asrc  = (const float*)d_in[6];
  const float* g1_adst  = (const float*)d_in[7];
  const float* g1_We    = (const float*)d_in[8];
  const float* g1_ae    = (const float*)d_in[9];
  const float* g1_b     = (const float*)d_in[10];
  const float* g2_Wl    = (const float*)d_in[11];
  const float* g2_bl    = (const float*)d_in[12];
  const float* g2_Wr    = (const float*)d_in[13];
  const float* g2_br    = (const float*)d_in[14];
  const float* g2_We    = (const float*)d_in[15];
  const float* g2_att   = (const float*)d_in[16];
  const float* g2_b     = (const float*)d_in[17];
  const float* a1_Wi    = (const float*)d_in[18];
  const float* a1_Wr    = (const float*)d_in[19];
  const float* a1_b     = (const float*)d_in[20];
  const float* a2_Wi    = (const float*)d_in[21];
  const float* a2_Wr    = (const float*)d_in[22];
  const float* a2_b     = (const float*)d_in[23];
  const float* gi_We    = (const float*)d_in[24];
  const float* gi_be    = (const float*)d_in[25];
  const float* m1       = (const float*)d_in[26];
  const float* mb1      = (const float*)d_in[27];
  const float* m2       = (const float*)d_in[28];
  const float* mb2      = (const float*)d_in[29];
  const float* m3       = (const float*)d_in[30];
  const float* mb3      = (const float*)d_in[31];
  const float* gi_eps   = (const float*)d_in[32];

  const int NN = in_sizes[0] / 64;     // 8000
  const int NE = in_sizes[1] / 2;      // 48000
  const int LE = in_sizes[3];
  const int* edst = ei + NE;           // edge_index row 1 (dst node per edge)

  // ---- workspace layout (float units, 256-float aligned blocks) ----
  float* wsf = (float*)d_ws;
  size_t off = 0;
  auto take = [&](size_t n) { float* p = wsf + off; off += (n + 255) & ~(size_t)255; return p; };
  float* h      = take((size_t)NE * 64);
  float* h2     = take((size_t)NE * 64);
  float* big    = take((size_t)NE * 256);   // xl -> xlv/xrv -> hWi -> z1
  float* xWe1   = take((size_t)NN * 4);
  float* xWe2   = take((size_t)NN * 64);
  float* xWeg   = take((size_t)NN * 64);
  float* deg    = take(NE);
  float* dinv   = take(NE);
  float* sl1    = take((size_t)NE * 4);
  float* sl2    = take((size_t)NE * 64);
  float* asrc   = take((size_t)NE * 4);
  float* adst   = take((size_t)NE * 4);
  float* segsum = take((size_t)NE * 4);
  float* aself  = take((size_t)NE * 4);
  float* alphaL = take((size_t)LE * 4);
  float* We1p   = take(64 * 4);
  _Float16* wh = (_Float16*)(wsf + off);
  size_t hoff = 0;
  auto takeh = [&](size_t n) { _Float16* p = wh + hoff; hoff += (n + 255) & ~(size_t)255; return p; };
  _Float16* g1_Wt = takeh(128 * 256);
  _Float16* wlT   = takeh(64 * 64);
  _Float16* wrT   = takeh(64 * 64);
  _Float16* we2T  = takeh(64 * 64);
  _Float16* wegT  = takeh(64 * 64);
  _Float16* wi1T  = takeh(64 * 64);
  _Float16* wr1T  = takeh(64 * 64);
  _Float16* wi2T  = takeh(64 * 64);
  _Float16* wr2T  = takeh(64 * 64);
  _Float16* m1T   = takeh(64 * 128);
  _Float16* m2T   = takeh(128 * 64);
  _Float16* m3T   = takeh(64 * 64);

  auto cdiv = [](int a, int b) { return (a + b - 1) / b; };
  auto conv = [&](const float* W, _Float16* Bt, int K, int N) {
    k_convw<<<cdiv(K * N, 256), 256, 0, stream>>>(W, Bt, K, N);
  };

  // weight prep
  conv(g1_W, g1_Wt, 128, 256);
  conv(g2_Wl, wlT, 64, 64);   conv(g2_Wr, wrT, 64, 64);
  conv(g2_We, we2T, 64, 64);  conv(gi_We, wegT, 64, 64);
  conv(a1_Wi, wi1T, 64, 64);  conv(a1_Wr, wr1T, 64, 64);
  conv(a2_Wi, wi2T, 64, 64);  conv(a2_Wr, wr2T, 64, 64);
  conv(m1, m1T, 64, 128);     conv(m2, m2T, 128, 64);   conv(m3, m3T, 64, 64);
  k_fold_atte<<<1, 256, 0, stream>>>(g1_We, g1_ae, We1p);

  // zero accumulators
  k_fill<<<cdiv(NE, 256), 256, 0, stream>>>(deg, 0.0f, NE);
  k_fill<<<cdiv(NE * 4, 256), 256, 0, stream>>>(sl1, 0.0f, NE * 4);
  k_fill<<<cdiv(NE * 64, 256), 256, 0, stream>>>(sl2, 0.0f, NE * 64);

  // per-node projections (WMMA)
  k_gemm<64, 64, 0><<<dim3(2, cdiv(NN, 64)), 128, 0, stream>>>(x, we2T, nullptr, xWe2, NN);
  k_gemm<64, 64, 0><<<dim3(2, cdiv(NN, 64)), 128, 0, stream>>>(x, wegT, nullptr, xWeg, NN);
  k_gemv<<<cdiv(NN * 4, 256), 256, 0, stream>>>(x, We1p, xWe1, NN, 64, 4);

  // GAT1 projection xl = edge_attr @ g1_W  [NE,256]  (WMMA)
  k_gemm<128, 256, 0><<<dim3(8, cdiv(NE, 64)), 128, 0, stream>>>(eattr, g1_Wt, nullptr, big, NE);
  k_attfold<<<cdiv(NE * 4, 256), 256, 0, stream>>>(big, g1_asrc, g1_adst, asrc, adst, NE);

  // line-graph degree + self-loop attrs
  k_deg_sl1<<<cdiv(LE, 256), 256, 0, stream>>>(ls, ld, edst, xWe1, deg, sl1, LE);
  k_sl2<<<cdiv(LE * 64, 256), 256, 0, stream>>>(ls, ld, edst, xWe2, sl2, LE);
  k_dinv<<<cdiv(NE, 256), 256, 0, stream>>>(deg, dinv, NE);

  // GAT1 softmax + aggregate -> h
  k_gat1_self<<<cdiv(NE * 4, 256), 256, 0, stream>>>(asrc, adst, sl1, deg, aself, segsum, NE);
  k_gat1_line<<<cdiv(LE * 4, 256), 256, 0, stream>>>(ls, ld, edst, asrc, adst, xWe1, alphaL, segsum, LE);
  k_gat1_init<<<cdiv(NE * 64, 256), 256, 0, stream>>>(aself, segsum, big, g1_b, h, NE);
  k_gat1_agg<<<cdiv(LE * 64, 256), 256, 0, stream>>>(ls, ld, alphaL, segsum, big, h, LE);

  // GATv2: xlv/xrv (WMMA, biased) into big
  float* xlv = big;
  float* xrv = big + (size_t)NE * 64;
  k_gemm<64, 64, 0><<<dim3(2, cdiv(NE, 64)), 128, 0, stream>>>(h, wlT, g2_bl, xlv, NE);
  k_gemm<64, 64, 0><<<dim3(2, cdiv(NE, 64)), 128, 0, stream>>>(h, wrT, g2_br, xrv, NE);
  k_gat2_self<<<cdiv(NE, 256), 256, 0, stream>>>(xlv, xrv, sl2, deg, g2_att, aself, segsum, NE);
  k_gat2_line<<<cdiv(LE, 256), 256, 0, stream>>>(ls, ld, edst, xlv, xrv, xWe2, g2_att, alphaL, segsum, LE);
  k_gat2_init<<<cdiv(NE * 64, 256), 256, 0, stream>>>(aself, segsum, xlv, g2_b, h2, NE);
  k_gat2_agg<<<cdiv(LE * 64, 256), 256, 0, stream>>>(ls, ld, alphaL, segsum, xlv, h2, LE);

  // ARMA layer 1: cur=h2 -> h
  k_gemm<64, 64, 0><<<dim3(2, cdiv(NE, 64)), 128, 0, stream>>>(h2, wi1T, nullptr, big, NE);
  k_gemm<64, 64, 0><<<dim3(2, cdiv(NE, 64)), 128, 0, stream>>>(h2, wr1T, a1_b, h, NE);
  k_arma_agg<<<cdiv(LE * 64, 256), 256, 0, stream>>>(ls, ld, dinv, big, h, LE);
  k_relu<<<cdiv(NE * 64, 256), 256, 0, stream>>>(h, NE * 64);

  // ARMA layer 2: cur=h -> h2
  k_gemm<64, 64, 0><<<dim3(2, cdiv(NE, 64)), 128, 0, stream>>>(h, wi2T, nullptr, big, NE);
  k_gemm<64, 64, 0><<<dim3(2, cdiv(NE, 64)), 128, 0, stream>>>(h, wr2T, a2_b, h2, NE);
  k_arma_agg<<<cdiv(LE * 64, 256), 256, 0, stream>>>(ls, ld, dinv, big, h2, LE);
  k_relu<<<cdiv(NE * 64, 256), 256, 0, stream>>>(h2, NE * 64);

  // GINE: z (in h) = (1+eps)*h2 + sum relu(h2[src] + xWeg[j] + be)
  k_zinit<<<cdiv(NE * 64, 256), 256, 0, stream>>>(h2, gi_eps, h, NE * 64);
  k_gine_agg<<<cdiv(LE * 64, 256), 256, 0, stream>>>(ls, ld, edst, h2, xWeg, gi_be, h, LE);

  // MLP head (WMMA with fused bias + leaky-relu)
  k_gemm<64, 128, 2><<<dim3(4, cdiv(NE, 64)), 128, 0, stream>>>(h, m1T, mb1, big, NE);
  k_gemm<128, 64, 2><<<dim3(2, cdiv(NE, 64)), 128, 0, stream>>>(big, m2T, mb2, h2, NE);
  k_gemm<64, 64, 0><<<dim3(2, cdiv(NE, 64)), 128, 0, stream>>>(h2, m3T, mb3, (float*)d_out, NE);
}
